// BitLinear_25314537243014
// MI455X (gfx1250) — 8-bit matrix operand order repaired and checked by reading, compiled, not run on hardware
//
#include <hip/hip_runtime.h>

typedef __attribute__((ext_vector_type(8))) int v8i;

#define EPSQ 1e-5f
#define K_DIM 1024
#define N_DIM 1024
#define M_DIM 32768

__global__ void wabs_partial_kernel(const float* __restrict__ w,
                                    float* __restrict__ partial) {
    __shared__ float smem[8];
    const int tid = threadIdx.x;
    const size_t base = (size_t)blockIdx.x * 1024 + (size_t)tid * 4;
    const float4 v = *(const float4*)(w + base);
    float s = fabsf(v.x) + fabsf(v.y) + fabsf(v.z) + fabsf(v.w);
#pragma unroll
    for (int off = 16; off > 0; off >>= 1) s += __shfl_xor(s, off, 32);
    if ((tid & 31) == 0) smem[tid >> 5] = s;
    __syncthreads();
    if (tid == 0) {
        float t = 0.f;
#pragma unroll
        for (int i = 0; i < 8; ++i) t += smem[i];
        partial[blockIdx.x] = t;
    }
}

__global__ void wmean_finalize_kernel(const float* __restrict__ partial,
                                      float* __restrict__ wscale) {
    __shared__ float smem[8];
    const int tid = threadIdx.x;
    float s = partial[tid] + partial[tid + 256] + partial[tid + 512] + partial[tid + 768];
#pragma unroll
    for (int off = 16; off > 0; off >>= 1) s += __shfl_xor(s, off, 32);
    if ((tid & 31) == 0) smem[tid >> 5] = s;
    __syncthreads();
    if (tid == 0) {
        float t = 0.f;
#pragma unroll
        for (int i = 0; i < 8; ++i) t += smem[i];
        wscale[0] = fmaxf(t * (1.0f / (1024.f * 1024.f)), EPSQ);
    }
}

__global__ void wquant_kernel(const float* __restrict__ w,
                              const float* __restrict__ wscale,
                              uint8_t* __restrict__ qw) {
    const size_t i = (size_t)blockIdx.x * 256 + threadIdx.x;
    const float4 v = ((const float4*)w)[i];
    const float sw = 1.0f / wscale[0];
    const int b0 = (int)fminf(fmaxf(rintf(v.x * sw), -1.f), 1.f);
    const int b1 = (int)fminf(fmaxf(rintf(v.y * sw), -1.f), 1.f);
    const int b2 = (int)fminf(fmaxf(rintf(v.z * sw), -1.f), 1.f);
    const int b3 = (int)fminf(fmaxf(rintf(v.w * sw), -1.f), 1.f);
    ((int*)qw)[i] = (b0 & 0xFF) | ((b1 & 0xFF) << 8) | ((b2 & 0xFF) << 16) | (b3 << 24);
}

__global__ void act_quant_kernel(const float* __restrict__ x,
                                 uint8_t* __restrict__ qx,
                                 float* __restrict__ rowscale) {
    const int lane = threadIdx.x & 31;
    const int wave = threadIdx.x >> 5;
    const int row  = blockIdx.x * 8 + wave;

    const float4* xr = (const float4*)(x + (size_t)row * K_DIM);
    float4 v[8];
    float amax = 0.f;
#pragma unroll
    for (int j = 0; j < 8; ++j) {
        v[j] = xr[j * 32 + lane];
        amax = fmaxf(amax, fmaxf(fmaxf(fabsf(v[j].x), fabsf(v[j].y)),
                                 fmaxf(fabsf(v[j].z), fabsf(v[j].w))));
    }
#pragma unroll
    for (int off = 16; off > 0; off >>= 1) amax = fmaxf(amax, __shfl_xor(amax, off, 32));
    amax = fmaxf(amax, EPSQ);
    const float qs = 127.f / amax;
    if (lane == 0) rowscale[row] = amax * (1.f / 127.f);

    int* qr = (int*)(qx + (size_t)row * K_DIM);
#pragma unroll
    for (int j = 0; j < 8; ++j) {
        const int b0 = (int)fminf(fmaxf(rintf(v[j].x * qs), -128.f), 127.f);
        const int b1 = (int)fminf(fmaxf(rintf(v[j].y * qs), -128.f), 127.f);
        const int b2 = (int)fminf(fmaxf(rintf(v[j].z * qs), -128.f), 127.f);
        const int b3 = (int)fminf(fmaxf(rintf(v[j].w * qs), -128.f), 127.f);
        qr[j * 32 + lane] = (b0 & 0xFF) | ((b1 & 0xFF) << 8) | ((b2 & 0xFF) << 16) | (b3 << 24);
    }
}

__device__ __forceinline__ v8i load_a_frag(const uint8_t* p) {
    const int2* q = (const int2*)p;
    const int2 t0 = q[0], t1 = q[1], t2 = q[4], t3 = q[5];
    v8i a;
    a[0] = t0.x; a[1] = t0.y; a[2] = t1.x; a[3] = t1.y;
    a[4] = t2.x; a[5] = t2.y; a[6] = t3.x; a[7] = t3.y;
    return a;
}
__device__ __forceinline__ v8i load_b_frag(const uint8_t* p) {
    const int4* q = (const int4*)p;
    const int4 t0 = q[0], t1 = q[2];
    v8i b;
    b[0] = t0.x; b[1] = t0.y; b[2] = t0.z; b[3] = t0.w;
    b[4] = t1.x; b[5] = t1.y; b[6] = t1.z; b[7] = t1.w;
    return b;
}

__global__ void __launch_bounds__(256)
bitlinear_gemm_kernel(const uint8_t* __restrict__ qx,
                      const uint8_t* __restrict__ qw,
                      const float* __restrict__ rowscale,
                      const float* __restrict__ wscale,
                      const float* __restrict__ bias,
                      float* __restrict__ out) {
    const int lane = threadIdx.x & 31;
    const int wave = threadIdx.x >> 5;
    const int lrow = lane & 15;
    const int half = lane >> 4;

    const int wave_m = wave & 3;
    const int wave_n = wave >> 2;
    const int m_base = blockIdx.y * 128 + wave_m * 32;
    const int n_base = blockIdx.x * 64 + wave_n * 32;

    v8i acc[2][2] = {};

    const uint8_t* pa0 = qx + (size_t)(m_base + lrow) * K_DIM + 16 * half;
    const uint8_t* pa1 = pa0 + (size_t)16 * K_DIM;
    const uint8_t* pb0 = qw + (size_t)(n_base + lrow) * K_DIM + 16 * half;
    const uint8_t* pb1 = pb0 + (size_t)16 * K_DIM;

#pragma unroll 2
    for (int k = 0; k < K_DIM; k += 64) {
        const v8i a0 = load_a_frag(pa0 + k);
        const v8i a1 = load_a_frag(pa1 + k);
        const v8i b0 = load_b_frag(pb0 + k);
        const v8i b1 = load_b_frag(pb1 + k);
        acc[0][0] = __builtin_amdgcn_wmma_i32_16x16x64_iu8(true, a0, true, b0, acc[0][0], false, false);
        acc[0][1] = __builtin_amdgcn_wmma_i32_16x16x64_iu8(true, a0, true, b1, acc[0][1], false, false);
        acc[1][0] = __builtin_amdgcn_wmma_i32_16x16x64_iu8(true, a1, true, b0, acc[1][0], false, false);
        acc[1][1] = __builtin_amdgcn_wmma_i32_16x16x64_iu8(true, a1, true, b1, acc[1][1], false, false);
    }

    const float wdq = wscale[0];
#pragma unroll
    for (int nt = 0; nt < 2; ++nt) {
        const int n = n_base + nt * 16 + lrow;
        const float bv = bias[n];
#pragma unroll
        for (int mt = 0; mt < 2; ++mt) {
#pragma unroll
            for (int r = 0; r < 8; ++r) {
                const int m = m_base + mt * 16 + r + 8 * half;
                out[(size_t)m * N_DIM + n] =
                    (float)acc[mt][nt][r] * rowscale[m] * wdq + bv;
            }
        }
    }
}

extern "C" void kernel_launch(void* const* d_in, const int* in_sizes, int n_in,
                              void* d_out, int out_size, void* d_ws, size_t ws_size,
                              hipStream_t stream) {
    const float* x    = (const float*)d_in[0];
    const float* w    = (const float*)d_in[1];
    const float* bias = (const float*)d_in[2];
    float* out = (float*)d_out;

    uint8_t* qx = (uint8_t*)d_ws;
    uint8_t* qw = qx + (size_t)M_DIM * K_DIM;
    float* rowscale = (float*)(qw + (size_t)N_DIM * K_DIM);
    float* partial  = rowscale + M_DIM;
    float* wscale   = partial + 1024;

    wabs_partial_kernel<<<1024, 256, 0, stream>>>(w, partial);
    wmean_finalize_kernel<<<1, 256, 0, stream>>>(partial, wscale);
    wquant_kernel<<<1024, 256, 0, stream>>>(w, wscale, qw);
    act_quant_kernel<<<M_DIM / 8, 256, 0, stream>>>(x, qx, rowscale);
    dim3 grid(N_DIM / 64, M_DIM / 128);
    bitlinear_gemm_kernel<<<grid, 256, 0, stream>>>(qx, qw, rowscale, wscale, bias, out);
}
